// dvs_gesture_snn_small_33122787787166
// MI455X (gfx1250) — matrix operand order repaired and checked by reading, not compiled, not run on hardware
//
#include <hip/hip_runtime.h>

typedef __attribute__((ext_vector_type(16))) _Float16 v16h;
typedef __attribute__((ext_vector_type(8)))  float    v8f;
typedef __attribute__((ext_vector_type(4)))  unsigned u32x4;
typedef __attribute__((ext_vector_type(8)))  unsigned u32x8;
typedef _Float16 half_t;

union H16 { v16h v; uint4 q[2]; };
union F8  { v8f  v; float4 f[2]; };

#define THRESH 1.0f
#define MIN_V  (-1.0f)

__device__ __forceinline__ float iaf_step(float& v, float u) {
    v += u;                                   // integrate
    float s = (v >= THRESH) ? 1.0f : 0.0f;    // SingleSpike
    v = (s != 0.0f) ? 0.0f : v;               // MembraneReset
    v = fmaxf(v, MIN_V);                      // min_v_mem clamp
    return s;
}

// padded im2col slab (halfs): 8 channels x 34 rows x 48 cols, zero borders
#define ROWQ 48
#define CHQ  (34 * ROWQ)      // 1632
#define SHQ  (8 * CHQ)        // 13056 halfs = ~26 KB

// K-slot permutation: lane halves hold slots (s, s+16) of each 32-slot chunk.  We pair
// k = q (ic 0..3) with k = q+36 (ic 4..7, same ky/kx), so hi-lo LDS offset
// delta is the constant 4*CHQ for every pair -> per-element addresses are
// pure immediate DS offsets.  Pair ordinal for fragment element (c,h) is
// q = 16c + h; real iff q < 36 (36 pairs = 72 real K slots).
__device__ constexpr int pair_off(int q) {   // lo-variant LDS offset (halfs)
    int ic = q / 9, pos = q % 9;
    return ic * CHQ + (pos / 3) * ROWQ + (pos % 3) + 7;
}

// ---------------------------------------------------------------------------
// k0: w2 [16,8,3,3] -> f16 [16][96] in slot order;  w3 [64,1024] -> f16 as-is
// ---------------------------------------------------------------------------
__global__ void k0_convert(const float* __restrict__ w2, const float* __restrict__ w3,
                           half_t* __restrict__ w2p, half_t* __restrict__ w3h) {
    int i = blockIdx.x * blockDim.x + threadIdx.x;
    if (i < 1536) {                       // w2p[oc*96 + slot]
        int oc = i / 96, s = i - oc * 96;
        int c = s >> 5, r = s & 31;
        int ord, hiHalf;
        if (r < 16)      { ord = r;      hiHalf = 0; }
        else             { ord = r - 16; hiHalf = 1; }
        int q = 16 * c + ord;
        int k = (q < 36) ? (q + (hiHalf ? 36 : 0)) : -1;
        w2p[i] = (k >= 0) ? (half_t)w2[oc * 72 + k] : (half_t)0.0f;
    }
    int j = i - 1536;
    if (j >= 0 && j < 65536) w3h[j] = (half_t)w3[j];   // natural [n][k] layout
}

// ---------------------------------------------------------------------------
// k1: conv1 (2->8, 3x3, s2, p1) + IAF1 scan over T + 2x2 avgpool
// ---------------------------------------------------------------------------
__global__ void __launch_bounds__(256)
k1_conv1_iaf_pool(const float* __restrict__ x, const float* __restrict__ w1,
                  half_t* __restrict__ s1) {
    int tid = blockIdx.x * blockDim.x + threadIdx.x;
    int b   = tid >> 13;
    int rem = tid & 8191;
    int oc  = rem >> 10;
    int ph  = (rem >> 5) & 31;
    int pw  = rem & 31;

    float wk[2][9];
#pragma unroll
    for (int ic = 0; ic < 2; ++ic)
#pragma unroll
        for (int j = 0; j < 9; ++j)
            wk[ic][j] = w1[(oc * 2 + ic) * 9 + j];

    float v[4] = {0.f, 0.f, 0.f, 0.f};
    const int iy0 = 4 * ph - 1, ix0 = 4 * pw - 1;

    for (int t = 0; t < 40; ++t) {
        const float* xb = x + ((size_t)(b * 40 + t) * 2) * 16384;
        if (t + 1 < 40) __builtin_prefetch(xb + 2 * 16384, 0, 0);  // global_prefetch_b8
        float patch[2][5][5];
#pragma unroll
        for (int ic = 0; ic < 2; ++ic) {
            const float* xc = xb + ic * 16384;
#pragma unroll
            for (int r = 0; r < 5; ++r) {
                int iy = iy0 + r;
#pragma unroll
                for (int cq = 0; cq < 5; ++cq) {
                    int ix = ix0 + cq;
                    bool ok = ((unsigned)iy < 128u) && ((unsigned)ix < 128u);
                    patch[ic][r][cq] = ok ? xc[iy * 128 + ix] : 0.f;
                }
            }
        }
        float acc = 0.f;
#pragma unroll
        for (int dy = 0; dy < 2; ++dy)
#pragma unroll
            for (int dx = 0; dx < 2; ++dx) {
                float s = 0.f;
#pragma unroll
                for (int ic = 0; ic < 2; ++ic)
#pragma unroll
                    for (int ky = 0; ky < 3; ++ky)
#pragma unroll
                        for (int kx = 0; kx < 3; ++kx)
                            s = fmaf(patch[ic][2 * dy + ky][2 * dx + kx],
                                     wk[ic][ky * 3 + kx], s);
                acc += iaf_step(v[dy * 2 + dx], s);
            }
        int bt = b * 40 + t;
        s1[((size_t)bt * 8 + oc) * 1024 + ph * 32 + pw] = (half_t)(acc * 0.25f);
    }
}

// ---------------------------------------------------------------------------
// k2: conv2 (8->16, 3x3, s2, p1) as implicit GEMM via WMMA f16
// padded spike slab staged in LDS via global_load_async_to_lds_b128 (ASYNCcnt)
// ---------------------------------------------------------------------------
__global__ void __launch_bounds__(512)
k2_conv2_wmma(const half_t* __restrict__ s1, const half_t* __restrict__ w2p,
              float* __restrict__ y2) {
    __shared__ __align__(16) half_t sh[SHQ];
    int bt = blockIdx.x;

    // 1) zero-fill padded slab (ds_store_b128)
    {
        uint4* z = (uint4*)sh;
        uint4 zz = {0u, 0u, 0u, 0u};
        for (int i = threadIdx.x; i < SHQ / 8; i += 512) z[i] = zz;
    }
    __syncthreads();

    // 2) async interior copy: 8 ch x 32 rows x 64B -> 1024 x b128, per-lane DMA
    {
        const half_t* gsrc = s1 + (size_t)bt * 8192;
        for (int j = threadIdx.x; j < 1024; j += 512) {
            int ic = j >> 7, iy = (j >> 2) & 31, q = j & 3;
            const uint4* g = (const uint4*)(gsrc + ic * 1024 + iy * 32 + q * 8);
            unsigned lofs =
                (unsigned)(uintptr_t)&sh[ic * CHQ + (iy + 1) * ROWQ + 8 + q * 8];
            asm volatile("global_load_async_to_lds_b128 %0, %1, off"
                         :: "v"(lofs), "v"(g) : "memory");
        }
        asm volatile("s_wait_asynccnt 0x0" ::: "memory");
    }
    __syncthreads();

    int lane = threadIdx.x & 31;
    int tile = threadIdx.x >> 5;   // 16 waves -> 16 spatial M-tiles
    int hi   = lane >> 4;
    int m    = lane & 15;

    int p    = tile * 16 + m;      // spatial position of this lane's A row
    int oy   = p >> 4, ox = p & 15;
    // single per-lane base; all gather elements are immediate DS offsets
    const half_t* ap = &sh[96 * oy + 2 * ox + (hi ? 4 * CHQ : 0)];

    v8f acc = {};
#pragma unroll
    for (int c = 0; c < 3; ++c) {
        // A fragment: q = 16c + h; real pairs q < 36, rest immediate zeros
        v16h a;
#pragma unroll
        for (int h = 0; h < 16; ++h) {
            const int q = 16 * c + h;
            a[h] = (q < 36) ? ap[pair_off(q)] : (half_t)0.0f;
        }
        // B fragment: 16 contiguous halfs -> 2 x b128
        H16 bf;
        const uint4* bp = (const uint4*)(w2p + m * 96 + 32 * c + 16 * hi);
        bf.q[0] = bp[0];
        bf.q[1] = bp[1];
        acc = __builtin_amdgcn_wmma_f32_16x16x32_f16(false, a, false, bf.v,
                                                     (short)0, acc, false, false);
    }
    // D: lane row m -> oc, 8 contiguous spatial positions -> 2 x b128 store
    F8 r; r.v = acc;
    float* drow = y2 + ((size_t)bt * 16 + m) * 256 + tile * 16 + 8 * hi;
    *(float4*)(drow + 0) = r.f[0];
    *(float4*)(drow + 4) = r.f[1];
}

// ---------------------------------------------------------------------------
// k3: IAF2 scan over T + 2x2 avgpool -> s2 [1000,1024] f16
// ---------------------------------------------------------------------------
__global__ void __launch_bounds__(256)
k3_iaf2_pool(const float* __restrict__ y2, half_t* __restrict__ s2) {
    int tid = blockIdx.x * blockDim.x + threadIdx.x;
    int b   = tid >> 10;
    int rem = tid & 1023;
    int oc  = rem >> 6;
    int ph  = (rem >> 3) & 7;
    int pw  = rem & 7;
    float v[4] = {0.f, 0.f, 0.f, 0.f};
    for (int t = 0; t < 40; ++t) {
        int bt = b * 40 + t;
        const float* yb = y2 + ((size_t)bt * 16 + oc) * 256;
        float acc = 0.f;
#pragma unroll
        for (int dy = 0; dy < 2; ++dy) {
            float2 u = *(const float2*)(yb + (2 * ph + dy) * 16 + 2 * pw);
            acc += iaf_step(v[dy * 2 + 0], u.x);
            acc += iaf_step(v[dy * 2 + 1], u.y);
        }
        s2[(size_t)bt * 1024 + oc * 64 + ph * 8 + pw] = (half_t)(acc * 0.25f);
    }
}

// ---------------------------------------------------------------------------
// k4: fc3 GEMM [1000,1024] x [1024,64]^T via WMMA f16, fp32 accumulate
// ---------------------------------------------------------------------------
__global__ void __launch_bounds__(256)
k4_fc3_wmma(const half_t* __restrict__ s2, const half_t* __restrict__ w3h,
            float* __restrict__ y3) {
    int wid = blockIdx.x * 8 + (threadIdx.x >> 5);
    if (wid >= 63 * 4) return;            // uniform per-wave exit
    int mt = wid >> 2, nt = wid & 3;
    int lane = threadIdx.x & 31;
    int hi = lane >> 4, m = lane & 15;
    int btA = mt * 16 + m; btA = (btA < 1000) ? btA : 999;

    const uint4* arow = (const uint4*)(s2 + (size_t)btA * 1024);
    const uint4* brow = (const uint4*)(w3h + (size_t)(nt * 16 + m) * 1024);

    v8f acc = {};
#pragma unroll 4
    for (int c = 0; c < 32; ++c) {
        H16 af, bf;
        af.q[0] = arow[4 * c + hi];           // halfs [32c+8hi,  +8)
        af.q[1] = arow[4 * c + hi + 2];       // halfs [32c+16+8hi, +8)
        bf.q[0] = brow[4 * c + hi];           // halfs [32c+8hi,  +8)
        bf.q[1] = brow[4 * c + hi + 2];       // halfs [32c+16+8hi, +8)
        acc = __builtin_amdgcn_wmma_f32_16x16x32_f16(false, af.v, false, bf.v,
                                                     (short)0, acc, false, false);
    }
#pragma unroll
    for (int vv = 0; vv < 8; ++vv) {
        int bt = mt * 16 + vv + 8 * hi;
        if (bt < 1000) y3[(size_t)bt * 64 + nt * 16 + m] = acc[vv];
    }
}

// ---------------------------------------------------------------------------
// k5: IAF3 scan + fc4 (64->11) + IAF4 -> out [25,40,11] fp32
// y3 slice (40x64 f32 = 10 KB, contiguous) staged to LDS with one TDM
// tensor_load_to_lds per wave, drained via s_wait_tensorcnt (TENSORcnt).
// ---------------------------------------------------------------------------
__global__ void __launch_bounds__(64)
k5_head(const float* __restrict__ y3, const float* __restrict__ w4,
        float* __restrict__ out) {
    __shared__ __align__(16) float shY[2560];   // 40 x 64 floats
    __shared__ float sp[64];
    int b = blockIdx.x;
    int f = threadIdx.x;

    {
        unsigned long long ga = (unsigned long long)(uintptr_t)(y3 + (size_t)b * 2560);
        unsigned lds = (unsigned)(uintptr_t)(&shY[0]);
        lds = __builtin_amdgcn_readfirstlane(lds);
        // D# group 0: count=1 | lds_addr | global_addr[56:0] | type=2
        u32x4 g0;
        g0[0] = 1u;
        g0[1] = lds;
        g0[2] = (unsigned)ga;
        g0[3] = ((unsigned)(ga >> 32) & 0x01FFFFFFu) | 0x80000000u;
        // D# group 1: data_size=3 (8B units); 1-D tensor/tile of 1280 units
        u32x8 g1;
        g1[0] = 0x00030000u;       // workgroup_mask=0, data_size=3
        g1[1] = 1280u << 16;       // tensor_dim0[15:0]
        g1[2] = 0u;                // tensor_dim0[31:16]=0, tensor_dim1=0
        g1[3] = 1280u << 16;       // tile_dim0 = 1280
        g1[4] = 0u;                // tile_dim1/2 unused
        g1[5] = 1280u;             // tensor_dim0_stride
        g1[6] = 0u;
        g1[7] = 0u;
        asm volatile("tensor_load_to_lds %0, %1" :: "s"(g0), "s"(g1) : "memory");
        __builtin_amdgcn_s_wait_tensorcnt(0);
    }
    __syncthreads();

    float v3 = 0.f, v4 = 0.f;
    for (int t = 0; t < 40; ++t) {
        int bt = b * 40 + t;
        sp[f] = iaf_step(v3, shY[t * 64 + f]);
        __syncthreads();
        if (f < 11) {
            float d = 0.f;
#pragma unroll
            for (int j = 0; j < 64; ++j) d = fmaf(sp[j], w4[f * 64 + j], d);
            out[bt * 11 + f] = iaf_step(v4, d);
        }
        __syncthreads();
    }
}

// ---------------------------------------------------------------------------
extern "C" void kernel_launch(void* const* d_in, const int* in_sizes, int n_in,
                              void* d_out, int out_size, void* d_ws, size_t ws_size,
                              hipStream_t stream) {
    const float* x  = (const float*)d_in[0];   // [25,40,2,128,128]
    const float* w1 = (const float*)d_in[1];   // [8,2,3,3]
    const float* w2 = (const float*)d_in[2];   // [16,8,3,3]
    const float* w3 = (const float*)d_in[3];   // [64,1024]
    const float* w4 = (const float*)d_in[4];   // [11,64]
    float* out = (float*)d_out;                // [25,40,11]

    char* p = (char*)d_ws;
    auto carve = [&](size_t bytes) -> char* {
        char* r = p;
        p += (bytes + 255) & ~(size_t)255;
        return r;
    };
    half_t* w2p = (half_t*)carve((size_t)1536 * 2);          // [16][96] slot order
    half_t* w3h = (half_t*)carve((size_t)65536 * 2);         // [64][1024]
    half_t* s1  = (half_t*)carve((size_t)1000 * 8192 * 2);   // pooled spikes L1
    float*  y2  = (float*) carve((size_t)1000 * 4096 * 4);   // conv2 pre-activation
    half_t* s2  = (half_t*)carve((size_t)1000 * 1024 * 2);   // pooled spikes L2
    float*  y3  = (float*) carve((size_t)1000 * 64 * 4);     // fc3 pre-activation

    k0_convert<<<(67072 + 255) / 256, 256, 0, stream>>>(w2, w3, w2p, w3h);
    k1_conv1_iaf_pool<<<800, 256, 0, stream>>>(x, w1, s1);
    k2_conv2_wmma<<<1000, 512, 0, stream>>>(s1, w2p, y2);
    k3_iaf2_pool<<<100, 256, 0, stream>>>(y2, s2);
    k4_fc3_wmma<<<32, 256, 0, stream>>>(s2, w3h, y3);
    k5_head<<<25, 64, 0, stream>>>(y3, w4, out);
}
